// CausalGAT_2697239461999
// MI455X (gfx1250) — matrix operand order repaired and checked by reading, not compiled, not run on hardware
//
#include <hip/hip_runtime.h>
#include <hip/hip_fp16.h>

// GAT forward for MI455X (gfx1250, wave32, WMMA).
// N=50000, E=800000, DIN=128, DH=64, H=4, DOUT=64.
//
// WMMA GEMMs consume f16 operands staged in workspace so that the
// CDNA5 16x16x32 fragment layouts map to contiguous b128 loads:
//   A row-major f16:   lane fragment = 2 contiguous 8-half runs
//   B transposed f16:  lane fragment = 2 contiguous 8-half runs

#define DIN  128
#define DH   64
#define NH   4
#define DOUT 64

typedef __attribute__((ext_vector_type(16))) _Float16 v16h;
typedef __attribute__((ext_vector_type(8)))  _Float16 v8h;
typedef __attribute__((ext_vector_type(8)))  float    v8f;

__device__ __forceinline__ float leaky(float x, float s) { return x >= 0.f ? x : s * x; }

// ---- WMMA fragment loaders (CDNA5 16-bit layouts, ISA 7.12.2) -------------
// A 16x32: lane (half,m): a[0..7] = Arow[k0+half*8 .. +8), a[8..15] = +16 more.
__device__ __forceinline__ v16h load_a_f16(const _Float16* __restrict__ A, int lda,
                                           int row0, int k0, int lane) {
  const int half = lane >> 4, m = lane & 15;
  const _Float16* p = A + (size_t)(row0 + m) * lda + k0 + half * 8;
  const v8h lo = *(const v8h*)(p);
  const v8h hi = *(const v8h*)(p + 16);
  v16h a;
#pragma unroll
  for (int i = 0; i < 8; ++i) { a[i] = lo[i]; a[8 + i] = hi[i]; }
  return a;
}

// B 32x16 from transposed weights Bt[Ncols][K]: lane (half,n):
// b[0..7] = Bt[n0+n][k0+half*8 .. +8), b[8..15] = +16 more
__device__ __forceinline__ v16h load_bt_f16(const _Float16* __restrict__ Bt, int K,
                                            int k0, int n0, int lane) {
  const int half = lane >> 4, n = lane & 15;
  const _Float16* p = Bt + (size_t)(n0 + n) * K + k0 + half * 8;
  const v8h lo = *(const v8h*)(p);
  const v8h hi = *(const v8h*)(p + 16);
  v16h b;
#pragma unroll
  for (int i = 0; i < 8; ++i) { b[i] = lo[i]; b[8 + i] = hi[i]; }
  return b;
}

__device__ __forceinline__ v8f wmma_f16(v16h a, v16h b, v8f c) {
  // (neg_a, A, neg_b, B, c_mod, C, reuse_a, reuse_b)
  return __builtin_amdgcn_wmma_f32_16x16x32_f16(false, a, false, b, (short)0, c,
                                                false, false);
}

// ---- order-preserving float<->uint map for atomic segment-max -------------
__device__ __forceinline__ unsigned enc_f32(float f) {
  unsigned u = __float_as_uint(f);
  return (u & 0x80000000u) ? ~u : (u | 0x80000000u);
}
__device__ __forceinline__ float dec_f32(unsigned k) {
  unsigned u = (k & 0x80000000u) ? (k & 0x7FFFFFFFu) : ~k;
  return __uint_as_float(u);
}

// ---- staging kernels ------------------------------------------------------
// vectorized f32 -> f16 (8 elements / thread)
__global__ void k_cvt_f16(const float* __restrict__ src, _Float16* __restrict__ dst,
                          long long count) {
  const long long t = ((long long)blockIdx.x * blockDim.x + threadIdx.x) * 8;
  if (t + 8 <= count) {
    const float4 a = *(const float4*)(src + t);
    const float4 b = *(const float4*)(src + t + 4);
    v8h o;
    o[0] = (_Float16)a.x; o[1] = (_Float16)a.y; o[2] = (_Float16)a.z; o[3] = (_Float16)a.w;
    o[4] = (_Float16)b.x; o[5] = (_Float16)b.y; o[6] = (_Float16)b.z; o[7] = (_Float16)b.w;
    *(v8h*)(dst + t) = o;
  } else {
    for (long long i = t; i < count; ++i) dst[i] = (_Float16)src[i];
  }
}

// W[K][Ncols] f32 -> Wt[Ncols][K] f16 (tiny matrices, coalesced writes)
__global__ void k_w_t_f16(const float* __restrict__ W, _Float16* __restrict__ Wt,
                          int K, int Ncols) {
  const int tid = blockIdx.x * blockDim.x + threadIdx.x;
  if (tid >= K * Ncols) return;
  const int n = tid / K, k = tid - n * K;
  Wt[tid] = (_Float16)W[(size_t)k * Ncols + n];
}

// ---- GEMM: out[N,64] = act(A[N,128] @ B[128,64] + bias) -------------------
// A: f16 row-major, Bt: f16 [64][128]. One wave per 16-row tile, 16 v_wmma.
// Optionally also emits an f16 copy of the output (feeds the next GEMM).
__global__ void k_lin128_64(const _Float16* __restrict__ A,
                            const _Float16* __restrict__ Bt,
                            const float* __restrict__ bias, float* __restrict__ out,
                            _Float16* __restrict__ outh, int n_tiles, float slope) {
  const int lane = threadIdx.x & 31;
  const int tile = blockIdx.x * (blockDim.x >> 5) + (threadIdx.x >> 5);
  if (tile >= n_tiles) return;
  const int row0 = tile * 16;
  v8f acc[4] = {};
#pragma unroll
  for (int kt = 0; kt < 4; ++kt) {
    const v16h a = load_a_f16(A, DIN, row0, kt * 32, lane);
#pragma unroll
    for (int ct = 0; ct < 4; ++ct) {
      const v16h b = load_bt_f16(Bt, DIN, kt * 32, ct * 16, lane);
      acc[ct] = wmma_f16(a, b, acc[ct]);
    }
  }
  const int half = lane >> 4, n = lane & 15;
#pragma unroll
  for (int ct = 0; ct < 4; ++ct) {
    const int col = ct * 16 + n;
    const float bv = bias[col];
#pragma unroll
    for (int r = 0; r < 8; ++r) {
      const float v = leaky(acc[ct][r] + bv, slope);
      const size_t idx = (size_t)(row0 + half * 8 + r) * DH + col;
      out[idx] = v;
      if (outh) outh[idx] = (_Float16)v;
    }
  }
}

// ---- per (row-tile, head): xs = h@W_src slice, xd = h@W_dst slice,
// a_src = <xs, att_src[head]>, a_dst = <xd, att_dst[head]> (wave32 xor-reduce)
__global__ void k_gat_proj(const _Float16* __restrict__ hh,
                           const _Float16* __restrict__ WtS,   // [256][64] f16
                           const _Float16* __restrict__ WtD,   // [256][64] f16
                           const float* __restrict__ att_s,
                           const float* __restrict__ att_d, float* __restrict__ xs,
                           float* __restrict__ a_src, float* __restrict__ a_dst,
                           int n_tiles) {
  const int lane = threadIdx.x & 31;
  const int wid  = blockIdx.x * (blockDim.x >> 5) + (threadIdx.x >> 5);
  const int tile = wid >> 2, head = wid & 3;
  if (tile >= n_tiles) return;
  const int row0 = tile * 16;
  v8f as_[4] = {};
  v8f ad_[4] = {};
#pragma unroll
  for (int kt = 0; kt < 2; ++kt) {
    const v16h a = load_a_f16(hh, DH, row0, kt * 32, lane);
#pragma unroll
    for (int ct = 0; ct < 4; ++ct) {
      const int n0 = head * DH + ct * 16;
      const v16h bs = load_bt_f16(WtS, DH, kt * 32, n0, lane);
      as_[ct] = wmma_f16(a, bs, as_[ct]);
      const v16h bd = load_bt_f16(WtD, DH, kt * 32, n0, lane);
      ad_[ct] = wmma_f16(a, bd, ad_[ct]);
    }
  }
  const int half = lane >> 4, n = lane & 15;
  float ps[8] = {}, pd[8] = {};
#pragma unroll
  for (int ct = 0; ct < 4; ++ct) {
    const int c = ct * 16 + n;
    const float asv = att_s[head * DH + c];
    const float adv = att_d[head * DH + c];
#pragma unroll
    for (int r = 0; r < 8; ++r) {
      const size_t row = (size_t)(row0 + half * 8 + r);
      xs[row * (NH * DH) + head * DH + c] = as_[ct][r];
      ps[r] += as_[ct][r] * asv;
      pd[r] += ad_[ct][r] * adv;
    }
  }
  // reduce over the 16 columns (lanes sharing the same half): xor masks 1..8
#pragma unroll
  for (int m = 1; m < 16; m <<= 1) {
#pragma unroll
    for (int r = 0; r < 8; ++r) {
      ps[r] += __shfl_xor(ps[r], m, 32);
      pd[r] += __shfl_xor(pd[r], m, 32);
    }
  }
  if (n == 0) {
#pragma unroll
    for (int r = 0; r < 8; ++r) {
      const int row = row0 + half * 8 + r;
      a_src[row * NH + head] = ps[r];
      a_dst[row * NH + head] = pd[r];
    }
  }
}

// ---- edge phase -----------------------------------------------------------
__global__ void k_edge_max(const int* __restrict__ ei, const float* __restrict__ a_src,
                           const float* __restrict__ a_dst,
                           unsigned* __restrict__ maxkey, int E) {
  const int tid = blockIdx.x * blockDim.x + threadIdx.x;
  if (tid >= E * NH) return;
  const int e = tid >> 2, hh = tid & 3;
  const int s = ei[e], d = ei[E + e];
  const float v = leaky(a_src[s * NH + hh] + a_dst[d * NH + hh], 0.2f);
  atomicMax(&maxkey[d * NH + hh], enc_f32(v));
}

__global__ void k_edge_exp(const int* __restrict__ ei, const float* __restrict__ a_src,
                           const float* __restrict__ a_dst,
                           const unsigned* __restrict__ maxkey,
                           float* __restrict__ den, float* __restrict__ exb, int E) {
  const int tid = blockIdx.x * blockDim.x + threadIdx.x;
  if (tid >= E * NH) return;
  const int e = tid >> 2, hh = tid & 3;
  const int s = ei[e], d = ei[E + e];
  const float v = leaky(a_src[s * NH + hh] + a_dst[d * NH + hh], 0.2f);
  float m = dec_f32(maxkey[d * NH + hh]);
  // reference: m = where(isfinite(m), m, 0)
  if ((__float_as_uint(m) & 0x7F800000u) == 0x7F800000u) m = 0.f;
  const float ex = __expf(v - m);
  exb[tid] = ex;
  atomicAdd(&den[d * NH + hh], ex);
}

__global__ void k_alpha(const int* __restrict__ ei, const float* __restrict__ exb,
                        const float* __restrict__ den, float* __restrict__ alpha_out,
                        int E) {
  const int tid = blockIdx.x * blockDim.x + threadIdx.x;
  if (tid >= E * NH) return;
  const int e = tid >> 2, hh = tid & 3;
  const int d = ei[E + e];
  alpha_out[tid] = exb[tid] / (den[d * NH + hh] + 1e-16f);
}

// msg scatter: thread per (edge, 4 channels); xs/agg are L2-resident (51 MB each)
__global__ void k_scatter(const int* __restrict__ ei, const float* __restrict__ xs,
                          const float* __restrict__ alpha, float* __restrict__ agg,
                          int E) {
  const long long tid = (long long)blockIdx.x * blockDim.x + threadIdx.x;
  if (tid >= (long long)E * (NH * DH / 4)) return;
  const int e    = (int)(tid >> 6);
  const int c0   = ((int)(tid & 63)) << 2;   // channel base, 4-aligned within a head
  const int head = c0 >> 6;
  const int s = ei[e], d = ei[E + e];
  const float  al = alpha[(size_t)e * NH + head];
  const float4 v  = *(const float4*)(xs + (size_t)s * (NH * DH) + c0);
  float* dst = agg + (size_t)d * (NH * DH) + c0;
  atomicAdd(dst + 0, v.x * al);
  atomicAdd(dst + 1, v.y * al);
  atomicAdd(dst + 2, v.z * al);
  atomicAdd(dst + 3, v.w * al);
}

// zh[N,128] f16 = [ leaky(h,0.01) | leaky(mean_heads(agg)+b_gat, 0.01) ]
__global__ void k_build_z(const float* __restrict__ h, const float* __restrict__ agg,
                          const float* __restrict__ b_gat, _Float16* __restrict__ zh,
                          int N) {
  const int tid = blockIdx.x * blockDim.x + threadIdx.x;
  if (tid >= N * 2 * DH) return;
  const int nrow = tid >> 7, c = tid & 127;
  float v;
  if (c < DH) {
    v = h[(size_t)nrow * DH + c];
  } else {
    const int c2 = c - DH;
    const float* a = agg + (size_t)nrow * (NH * DH) + c2;
    v = 0.25f * (a[0] + a[DH] + a[2 * DH] + a[3 * DH]) + b_gat[c2];
  }
  zh[(size_t)nrow * (2 * DH) + c] = (_Float16)leaky(v, 0.01f);
}

extern "C" void kernel_launch(void* const* d_in, const int* in_sizes, int n_in,
                              void* d_out, int out_size, void* d_ws, size_t ws_size,
                              hipStream_t stream) {
  const float* X     = (const float*)d_in[0];
  const int*   ei    = (const int*)  d_in[1];
  const float* W_lin = (const float*)d_in[2];
  const float* b_lin = (const float*)d_in[3];
  const float* W_src = (const float*)d_in[4];
  const float* W_dst = (const float*)d_in[5];
  const float* att_s = (const float*)d_in[6];
  const float* att_d = (const float*)d_in[7];
  const float* b_gat = (const float*)d_in[8];
  const float* W_fin = (const float*)d_in[9];
  const float* b_fin = (const float*)d_in[10];

  const int N = in_sizes[0] / DIN;   // 50000
  const int E = in_sizes[1] / 2;     // 800000

  // ---- workspace carve-up: f32 region, then 16B-aligned f16 region --------
  float* ws = (float*)d_ws;
  size_t off = 0;
  float*    h      = ws + off; off += (size_t)N * DH;          // 12.8 MB
  float*    xs     = ws + off; off += (size_t)N * NH * DH;     // 51.2 MB
  float*    a_s    = ws + off; off += (size_t)N * NH;
  float*    a_d    = ws + off; off += (size_t)N * NH;
  const size_t z0  = off;
  unsigned* maxkey = (unsigned*)(ws + off); off += (size_t)N * NH;
  float*    den    = ws + off; off += (size_t)N * NH;
  float*    agg    = ws + off; off += (size_t)N * NH * DH;     // 51.2 MB
  const size_t zeroFloats = off - z0;                          // maxkey+den+agg
  float*    exb    = ws + off; off += (size_t)E * NH;          // 12.8 MB

  off = (off + 3) & ~(size_t)3;                                // 16B align
  _Float16* hp = (_Float16*)(ws + off);
  size_t hoff = 0;
  _Float16* Xh    = hp + hoff; hoff += (size_t)N * DIN;        // 12.8 MB
  _Float16* hh    = hp + hoff; hoff += (size_t)N * DH;         //  6.4 MB
  _Float16* zh    = hp + hoff; hoff += (size_t)N * 2 * DH;     // 12.8 MB
  _Float16* WtLin = hp + hoff; hoff += (size_t)DIN * DH;       // [64][128]
  _Float16* WtSrc = hp + hoff; hoff += (size_t)DH * NH * DH;   // [256][64]
  _Float16* WtDst = hp + hoff; hoff += (size_t)DH * NH * DH;   // [256][64]
  _Float16* WtFin = hp + hoff; hoff += (size_t)(2 * DH) * DOUT;// [64][128]

  float* out_main  = (float*)d_out;                    // [N, DOUT]
  float* out_alpha = (float*)d_out + (size_t)N * DOUT; // [E, H]

  // zero segment-max keys (0 decodes to -NaN -> caught by the finite check),
  // softmax denominators and the aggregation buffer.
  hipMemsetAsync((void*)maxkey, 0, zeroFloats * sizeof(float), stream);

  // ---- stage f16 operands -------------------------------------------------
  const long long xcnt = (long long)N * DIN;
  k_cvt_f16<<<(int)((xcnt / 8 + 255) / 256), 256, 0, stream>>>(X, Xh, xcnt);
  k_w_t_f16<<<(DIN * DH + 255) / 256, 256, 0, stream>>>(W_lin, WtLin, DIN, DH);
  k_w_t_f16<<<(DH * NH * DH + 255) / 256, 256, 0, stream>>>(W_src, WtSrc, DH, NH * DH);
  k_w_t_f16<<<(DH * NH * DH + 255) / 256, 256, 0, stream>>>(W_dst, WtDst, DH, NH * DH);
  k_w_t_f16<<<(2 * DH * DOUT + 255) / 256, 256, 0, stream>>>(W_fin, WtFin, 2 * DH, DOUT);

  const int tiles = N / 16;  // 3125, N divisible by 16
  // 1) h = leaky(X @ W_lin + b_lin, 0.01)   (emits f32 h + f16 hh)
  k_lin128_64<<<(tiles + 3) / 4, 128, 0, stream>>>(Xh, WtLin, b_lin, h, hh, tiles,
                                                   0.01f);
  // 2) xs, a_src, a_dst (dual WMMA GEMM + attention dots)
  k_gat_proj<<<tiles, 128, 0, stream>>>(hh, WtSrc, WtDst, att_s, att_d, xs, a_s, a_d,
                                        tiles);
  // 3..5) edge softmax
  const int eh = E * NH;
  k_edge_max<<<(eh + 255) / 256, 256, 0, stream>>>(ei, a_s, a_d, maxkey, E);
  k_edge_exp<<<(eh + 255) / 256, 256, 0, stream>>>(ei, a_s, a_d, maxkey, den, exb, E);
  k_alpha<<<(eh + 255) / 256, 256, 0, stream>>>(ei, exb, den, out_alpha, E);
  // 6) weighted scatter-sum of messages
  const long long sc = (long long)E * (NH * DH / 4);
  k_scatter<<<(int)((sc + 255) / 256), 256, 0, stream>>>(ei, xs, out_alpha, agg, E);
  // 7) concat + mean-over-heads + bias + leaky -> f16 z
  const int nz = N * 2 * DH;
  k_build_z<<<(nz + 255) / 256, 256, 0, stream>>>(h, agg, b_gat, zh, N);
  // 8) out = z @ W_fin + b_fin (slope=1 => identity activation)
  k_lin128_64<<<(tiles + 3) / 4, 128, 0, stream>>>(zh, WtFin, b_fin, out_main, nullptr,
                                                   tiles, 1.0f);
}
